// feature_extraction_13915694039526
// MI455X (gfx1250) — compile-verified, matrix operand order checked by reading, not run on hardware
//
#include <hip/hip_runtime.h>

#define NB 8
#define NPTS 4096
#define KNN_K 16
#define CHUNK 256
#define DB_NT 8
#define DB_COLS (DB_NT * 16)

typedef __attribute__((ext_vector_type(16))) __bf16 v16bf;
typedef __attribute__((ext_vector_type(8)))  __bf16 v8bf;
typedef __attribute__((ext_vector_type(8)))  float  v8f;
typedef __attribute__((ext_vector_type(4)))  unsigned int v4u;
typedef __attribute__((ext_vector_type(8)))  int v8i;
typedef __attribute__((ext_vector_type(4)))  int v4i;

// ---------------------------------------------------------------------
// Fragment load: operand staged K-contiguous (padded to mult of 32).
// Lane's halfs 0..7 are K=klo..klo+7, halfs 8..15 are K+16..K+23
// (klo = 8 for lanes 16..31): exactly two 16-byte vector loads.
// ---------------------------------------------------------------------
__device__ __forceinline__ v16bf frag16(const __bf16* p, int klo) {
  v8bf lo = *(const v8bf*)(p + klo);
  v8bf hi = *(const v8bf*)(p + 16 + klo);
  return __builtin_shufflevector(lo, hi, 0, 1, 2, 3, 4, 5, 6, 7, 8, 9, 10, 11,
                                 12, 13, 14, 15);
}

__device__ __forceinline__ v8f wmma_bf16(v16bf a, v16bf b, v8f c) {
  return __builtin_amdgcn_wmma_f32_16x16x32_bf16(false, a, false, b, (short)0, c,
                                                 false, false);
}

// ---------------------------------------------------------------------
// TDM: 1-D contiguous copy of `elems` bf16 from global -> LDS.
// D# per CDNA5 ISA: g0 = {count=1, lds_addr, gaddr,
// type=2}, g1 = {data_size=2B, tensor_dim0=tile_dim0=elems, dim1=1}.
// 6-arg builtin (clang-23 toolchain). Issue from one wave.
// ---------------------------------------------------------------------
__device__ __forceinline__ void tdm_copy_1d(unsigned ldsOff, const void* gsrc,
                                            unsigned elems) {
  unsigned long long ga = (unsigned long long)gsrc;
  v4u g0;
  g0[0] = 1u;                                               // count = 1
  g0[1] = ldsOff;                                           // LDS byte addr
  g0[2] = (unsigned)ga;                                     // gaddr lo
  g0[3] = (unsigned)((ga >> 32) & 0x1FFFFFFu) | (2u << 30); // gaddr hi | type=2
  v8i g1;
  g1[0] = 0x00010000;                           // data_size = 1 -> 2 bytes
  g1[1] = (int)((elems & 0xFFFFu) << 16);       // tensor_dim0[15:0]
  g1[2] = (int)((elems >> 16) & 0xFFFFu) | (1 << 16);  // dim0 hi | tensor_dim1=1
  g1[3] = (int)((elems & 0xFFFFu) << 16);       // tile_dim0 = elems (<65536)
  g1[4] = 0;                                    // tile_dim1 = 0 (1-D)
  g1[5] = (int)elems;                           // tensor_dim0_stride
  g1[6] = 0;
  g1[7] = 0;
  v4i z4 = {0, 0, 0, 0};
  v8i z8 = {0, 0, 0, 0, 0, 0, 0, 0};
  __builtin_amdgcn_tensor_load_to_lds(g0, g1, z4, z4, z8, 0);
}

// =====================================================================
// weight prep: W [O,Cin] f32 -> Wb [Opad,Kpad] bf16, zero padded
// =====================================================================
__global__ __launch_bounds__(256) void k_wprep(const float* __restrict__ W,
                                               int O, int Opad, int Cin,
                                               int Kpad,
                                               __bf16* __restrict__ out) {
  int t = blockIdx.x * 256 + threadIdx.x;
  if (t >= Opad * Kpad) return;
  int m = t / Kpad, k = t % Kpad;
  out[t] = (m < O && k < Cin) ? (__bf16)W[(size_t)m * Cin + k] : (__bf16)0.f;
}

// =====================================================================
// activation mirror: x [C,N] f32 (channel-major) -> xm [N][KP] bf16
// (point-major, zero padded). LDS-tiled transpose for coalescing.
// =====================================================================
__global__ __launch_bounds__(256) void k_x2bf(const float* __restrict__ x,
                                              long bStride, int C, int KP,
                                              __bf16* __restrict__ xm) {
  __shared__ __align__(16) __bf16 S[64][480];
  int b = blockIdx.y, n0 = blockIdx.x * 64;
  int tid = threadIdx.x;
  const float* xb = x + (size_t)b * bStride;
  for (int e = tid; e < 64 * KP; e += 256) {
    int col = e & 63, k = e >> 6;  // wave-uniform k
    S[col][k] = (k < C) ? (__bf16)xb[(size_t)k * NPTS + n0 + col] : (__bf16)0.f;
  }
  __syncthreads();
  int segs = KP / 8;
  for (int e = tid; e < 64 * segs; e += 256) {
    int seg = e % segs, col = e / segs;
    *(v8bf*)(xm + ((size_t)b * NPTS + n0 + col) * KP + seg * 8) =
        *(const v8bf*)&S[col][seg * 8];
  }
}

// =====================================================================
// c1: [B,3,N] -> relu(W x + b), 24 out channels (tiny; scalar VALU)
// =====================================================================
__global__ __launch_bounds__(256) void k_conv_c1(const float* __restrict__ xin,
                                                 const float* __restrict__ W,
                                                 const float* __restrict__ bias,
                                                 float* __restrict__ out,
                                                 long oStride) {
  int t = blockIdx.x * 256 + threadIdx.x;
  int b = t / NPTS, n = t % NPTS;
  float x0 = xin[((size_t)b * 3 + 0) * NPTS + n];
  float x1 = xin[((size_t)b * 3 + 1) * NPTS + n];
  float x2 = xin[((size_t)b * 3 + 2) * NPTS + n];
#pragma unroll
  for (int o = 0; o < 24; ++o) {
    float v = W[o * 3 + 0] * x0 + W[o * 3 + 1] * x1 + W[o * 3 + 2] * x2 + bias[o];
    out[(size_t)b * oStride + (size_t)o * NPTS + n] = fmaxf(v, 0.f);
  }
}

// =====================================================================
// squared norms per point
// =====================================================================
__global__ __launch_bounds__(256) void k_sqnorm(const float* __restrict__ x,
                                                long bStride, int C,
                                                float* __restrict__ sq) {
  int t = blockIdx.x * 256 + threadIdx.x;
  int b = t / NPTS, n = t % NPTS;
  const float* xb = x + (size_t)b * bStride;
  float s = 0.f;
  for (int c = 0; c < C; ++c) {
    float v = xb[(size_t)c * NPTS + n];
    s += v * v;
  }
  sq[(size_t)b * NPTS + n] = s;
}

// =====================================================================
// kNN: chunk columns staged by TDM (contiguous rows of the bf16 mirror),
// Gram via WMMA, running top-17 per row with early break.
// =====================================================================
template <int C, int KP>
__global__ __launch_bounds__(512) void k_knn(const __bf16* __restrict__ xm,
                                             const float* __restrict__ sq,
                                             int* __restrict__ idxOut) {
  constexpr int KSTEPS = KP / 32;
  __shared__ __align__(16) __bf16 XT[CHUNK][KP];
  __shared__ float dist[16][CHUNK];
  __shared__ float rsq[16];
  __shared__ float topD[16][17];
  __shared__ int topI[16][17];

  int b = blockIdx.y, row0 = blockIdx.x * 16;
  int tid = threadIdx.x, wave = tid >> 5, lane = tid & 31;
  int klo = (lane & 16) ? 8 : 0;
  const __bf16* xmb = xm + (size_t)b * NPTS * KP;
  const float* sqb = sq + (size_t)b * NPTS;

  if (tid < 16 * 17) topD[tid / 17][tid % 17] = __builtin_inff();
  if (tid < 16) rsq[tid] = sqb[row0 + tid];
  __syncthreads();

  // A fragments straight from the mirror (rows are K-contiguous).
  v16bf afr[KSTEPS];
#pragma unroll
  for (int ks = 0; ks < KSTEPS; ++ks)
    afr[ks] = frag16(xmb + (size_t)(row0 + (lane & 15)) * KP + ks * 32, klo);

  for (int ch = 0; ch < NPTS / CHUNK; ++ch) {
    int cbase = ch * CHUNK;
    // ---- TDM: copy CHUNK contiguous mirror rows into LDS ----
    if (tid < 32) {
      tdm_copy_1d((unsigned)(uintptr_t)&XT[0][0],
                  xmb + (size_t)cbase * KP, CHUNK * KP);
      __builtin_amdgcn_s_wait_tensorcnt(0);
    }
    __syncthreads();
    // ---- Gram tile per wave ----
    {
      int col0 = wave * 16;
      v8f acc = {0.f, 0.f, 0.f, 0.f, 0.f, 0.f, 0.f, 0.f};
#pragma unroll
      for (int ks = 0; ks < KSTEPS; ++ks)
        acc = wmma_bf16(afr[ks], frag16(&XT[col0 + (lane & 15)][ks * 32], klo),
                        acc);
      int colA = cbase + col0 + (lane & 15);
      float sqc = sqb[colA];
#pragma unroll
      for (int v = 0; v < 8; ++v) {
        int m = v + ((lane & 16) ? 8 : 0);
        float d = rsq[m] + sqc - 2.f * acc[v];
        if (colA == row0 + m) d = -__builtin_inff();
        dist[m][col0 + (lane & 15)] = d;
      }
    }
    __syncthreads();
    // ---- running top-17 insertion, wave w owns row w ----
    {
      volatile float* drow = &dist[wave][0];
      volatile float* tD = &topD[wave][0];
      for (int pass = 0; pass < 17; ++pass) {
        float mv = __builtin_inff();
        int mi = -1;
#pragma unroll
        for (int c = lane; c < CHUNK; c += 32) {
          float v = drow[c];
          if (v < mv) { mv = v; mi = c; }
        }
#pragma unroll
        for (int off = 16; off > 0; off >>= 1) {
          float ov = __shfl_xor(mv, off, 32);
          int oi = __shfl_xor(mi, off, 32);
          if (ov < mv || (ov == mv && oi >= 0 && (mi < 0 || oi < mi))) {
            mv = ov; mi = oi;
          }
        }
        if (mv >= tD[16]) break;
        if (lane == 0) {
          int pos = 16;
          while (pos > 0 && tD[pos - 1] > mv) {
            tD[pos] = tD[pos - 1];
            topI[wave][pos] = topI[wave][pos - 1];
            --pos;
          }
          tD[pos] = mv;
          topI[wave][pos] = cbase + mi;
          drow[mi] = __builtin_inff();
        }
      }
    }
    __syncthreads();
  }
  if (lane == 0) {
    int row = row0 + wave;
#pragma unroll
    for (int k = 0; k < KNN_K; ++k)
      idxOut[((size_t)b * KNN_K + k) * NPTS + row] = topI[wave][k + 1];
  }
}

// =====================================================================
// transition 1x1 conv: TDM stages the whole [64][KP] activation tile
// from the L-mirror; weights pre-padded bf16 [48][KP]. No masks.
// =====================================================================
template <int KP>
__global__ __launch_bounds__(256) void k_conv1x1(const __bf16* __restrict__ xm,
                                                 const __bf16* __restrict__ Wb,
                                                 const float* __restrict__ bias,
                                                 float* __restrict__ out,
                                                 long oStride) {
  constexpr int KSTEPS = KP / 32;
  __shared__ __align__(16) __bf16 XT[64][KP];
  int b = blockIdx.y, n0 = blockIdx.x * 64;
  int tid = threadIdx.x, wave = tid >> 5, lane = tid & 31;
  int klo = (lane & 16) ? 8 : 0;

  if (tid < 32) {
    tdm_copy_1d((unsigned)(uintptr_t)&XT[0][0],
                xm + ((size_t)b * NPTS + n0) * KP, 64 * KP);
    __builtin_amdgcn_s_wait_tensorcnt(0);
  }
  __syncthreads();

  for (int t = wave; t < 12; t += 8) {  // 3 M-tiles x 4 N-tiles
    int m0 = (t >> 2) * 16, c0 = (t & 3) * 16;
    const __bf16* wrow = Wb + (size_t)(m0 + (lane & 15)) * KP;
    v8f acc = {0.f, 0.f, 0.f, 0.f, 0.f, 0.f, 0.f, 0.f};
#pragma unroll
    for (int ks = 0; ks < KSTEPS; ++ks)
      acc = wmma_bf16(frag16(wrow + ks * 32, klo),
                      frag16(&XT[c0 + (lane & 15)][ks * 32], klo), acc);
    int nn = n0 + c0 + (lane & 15);
#pragma unroll
    for (int v = 0; v < 8; ++v) {
      int m = m0 + v + ((lane & 16) ? 8 : 0);  // < 48 always
      out[(size_t)b * oStride + (size_t)m * NPTS + nn] =
          fmaxf(acc[v] + bias[m], 0.f);
    }
  }
}

// =====================================================================
// dense block. Tt[col][row] in REFERENCE channel order (newest first):
// rows [0,24)=t3, [24,48)=t2, [48,72)=t1, [72,72+2C)=y. GEMM s reads
// rows base+k, base = 72-24*(s-1) -> fragments are 2x ds_load_b128.
// Gather uses v8bf loads from the point-major mirror.
// =====================================================================
template <int KP, int RD>
__device__ __forceinline__ void dense_gemm(__bf16 (*Tt)[RD], int srcBase,
                                           const __bf16* __restrict__ Wb,
                                           const float* __restrict__ bias,
                                           int dstRow, bool doRelu, int wave,
                                           int lane, int klo) {
  constexpr int KSTEPS = KP / 32;
  for (int t = wave; t < 16; t += 8) {  // 2 M-tiles x 8 N-tiles
    int m0 = (t >> 3) * 16, col0 = (t & 7) * 16;
    const __bf16* wrow = Wb + (size_t)(m0 + (lane & 15)) * KP;  // Opad = 32
    v8f acc = {0.f, 0.f, 0.f, 0.f, 0.f, 0.f, 0.f, 0.f};
#pragma unroll
    for (int ks = 0; ks < KSTEPS; ++ks)
      acc = wmma_bf16(
          frag16(wrow + ks * 32, klo),
          frag16(&Tt[col0 + (lane & 15)][srcBase + ks * 32], klo), acc);
    int cc = col0 + (lane & 15);
#pragma unroll
    for (int v = 0; v < 8; ++v) {
      int m = m0 + v + ((lane & 16) ? 8 : 0);
      if (m < 24) {
        float r = acc[v] + bias[m];
        if (doRelu) r = fmaxf(r, 0.f);
        Tt[cc][dstRow + m] = (__bf16)r;
      }
    }
  }
}

template <int C, int KPX>
__global__ __launch_bounds__(256) void k_dense(
    const __bf16* __restrict__ xm, const int* __restrict__ idx,
    const __bf16* __restrict__ W1b, const float* __restrict__ b1,
    const __bf16* __restrict__ W2b, const float* __restrict__ b2,
    const __bf16* __restrict__ W3b, const float* __restrict__ b3,
    float* __restrict__ out, long oStride) {
  constexpr int C2 = 2 * C;
  constexpr int KP1 = ((C2 + 31) / 32) * 32;
  constexpr int KP2 = ((C2 + 24 + 31) / 32) * 32;
  constexpr int KP3 = ((C2 + 48 + 31) / 32) * 32;
  constexpr int RDa = (24 + KP3 > 48 + KP2) ? 24 + KP3 : 48 + KP2;
  constexpr int RD = (RDa > 72 + KP1) ? RDa : 72 + KP1;  // mult of 8
  __shared__ __align__(16) __bf16 Tt[DB_COLS][RD];

  int b = blockIdx.y, n0 = blockIdx.x * DB_NT;
  int tid = threadIdx.x, wave = tid >> 5, lane = tid & 31;
  int klo = (lane & 16) ? 8 : 0;
  const __bf16* xmb = xm + (size_t)b * NPTS * KPX;
  const int* idxb = idx + (size_t)b * KNN_K * NPTS;

  // ---- phase A: gather edge features into rows [72, 72+2C) ----
  constexpr int SEGS = C / 8;
  for (int e = tid; e < DB_COLS * SEGS; e += 256) {
    int seg = e % SEGS, col = e / SEGS;
    int p = col >> 4, k = col & 15, n = n0 + p;
    int j = idxb[(size_t)k * NPTS + n];
    v8bf ctr = *(const v8bf*)(xmb + (size_t)n * KPX + seg * 8);
    v8bf nbr = *(const v8bf*)(xmb + (size_t)j * KPX + seg * 8);
#pragma unroll
    for (int h = 0; h < 8; ++h) {
      Tt[col][72 + seg * 8 + h] = ctr[h];
      Tt[col][72 + C + seg * 8 + h] = (__bf16)((float)nbr[h] - (float)ctr[h]);
    }
  }
  for (int e = tid; e < (RD - (72 + C2)) * DB_COLS; e += 256)
    Tt[e & 127][72 + C2 + (e >> 7)] = (__bf16)0.f;  // zero K padding rows
  __syncthreads();

  dense_gemm<KP1, RD>(Tt, 72, W1b, b1, 48, true, wave, lane, klo);
  __syncthreads();
  dense_gemm<KP2, RD>(Tt, 48, W2b, b2, 24, true, wave, lane, klo);
  __syncthreads();
  dense_gemm<KP3, RD>(Tt, 24, W3b, b3, 0, false, wave, lane, klo);
  __syncthreads();

  // ---- phase E: rows already in reference order -> row == out channel ----
  constexpr int CO = C2 + 72;
  for (int e = tid; e < CO * DB_NT; e += 256) {
    int oc = e / DB_NT, p = e % DB_NT;
    float mv = -__builtin_inff();
#pragma unroll
    for (int k = 0; k < 16; ++k) mv = fmaxf(mv, (float)Tt[p * 16 + k][oc]);
    out[(size_t)b * oStride + (size_t)oc * NPTS + (n0 + p)] = mv;
  }
}

// =====================================================================
// per-batch contiguous copy (concat history blocks)
// =====================================================================
__global__ void k_copy(const float* __restrict__ src, long sStride,
                       float* __restrict__ dst, long dStride, long cnt) {
  long total = (long)NB * cnt;
  for (long e = (long)blockIdx.x * blockDim.x + threadIdx.x; e < total;
       e += (long)gridDim.x * blockDim.x) {
    long b = e / cnt, r = e % cnt;
    dst[b * dStride + r] = src[b * sStride + r];
  }
}

// =====================================================================
extern "C" void kernel_launch(void* const* d_in, const int* in_sizes, int n_in,
                              void* d_out, int out_size, void* d_ws,
                              size_t ws_size, hipStream_t stream) {
  const float* xin = (const float*)d_in[0];
  const float* c1w = (const float*)d_in[1]; const float* c1b = (const float*)d_in[2];
  const float* cw[3] = {(const float*)d_in[3], (const float*)d_in[5],
                        (const float*)d_in[7]};
  const float* cb[3] = {(const float*)d_in[4], (const float*)d_in[6],
                        (const float*)d_in[8]};
  const float *dW[4][3], *dB[4][3];
  for (int blk = 0; blk < 4; ++blk)
    for (int l = 0; l < 3; ++l) {
      dW[blk][l] = (const float*)d_in[9 + blk * 6 + l * 2];
      dB[blk][l] = (const float*)d_in[9 + blk * 6 + l * 2 + 1];
    }

  const long s144 = 144L * NPTS, s312 = 312L * NPTS, s480 = 480L * NPTS;
  const long s48 = 48L * NPTS, s648 = 648L * NPTS;

  char* base = (char*)d_ws;
  size_t off = 0;
  auto allocB = [&](size_t bytes) {
    off = (off + 255) & ~(size_t)255;
    char* p = base + off;
    off += bytes;
    return p;
  };
  float* L1 = (float*)allocB((size_t)NB * s144 * 4);
  float* L2 = (float*)allocB((size_t)NB * s312 * 4);
  float* L3 = (float*)allocB((size_t)NB * s480 * 4);
  float* X  = (float*)allocB((size_t)NB * s48 * 4);
  float* SQ = (float*)allocB((size_t)NB * NPTS * 4);
  int* IDX  = (int*)allocB((size_t)NB * KNN_K * NPTS * 4);
  // bf16 point-major mirrors
  __bf16* M0  = (__bf16*)allocB((size_t)NB * NPTS * 32 * 2);
  __bf16* MX  = (__bf16*)allocB((size_t)NB * NPTS * 64 * 2);
  __bf16* ML1 = (__bf16*)allocB((size_t)NB * NPTS * 160 * 2);
  __bf16* ML2 = (__bf16*)allocB((size_t)NB * NPTS * 320 * 2);
  __bf16* ML3 = (__bf16*)allocB((size_t)NB * NPTS * 480 * 2);
  // padded bf16 weights
  const int cKP[3] = {160, 320, 480};
  __bf16* cWb[3];
  for (int i = 0; i < 3; ++i) cWb[i] = (__bf16*)allocB((size_t)48 * cKP[i] * 2);
  const int dKP[4][3] = {{64, 96, 96}, {96, 128, 160}, {96, 128, 160},
                         {96, 128, 160}};
  __bf16* dWb[4][3];
  for (int blk = 0; blk < 4; ++blk)
    for (int l = 0; l < 3; ++l)
      dWb[blk][l] = (__bf16*)allocB((size_t)32 * dKP[blk][l] * 2);
  float* out = (float*)d_out;
  float* l0 = L1 + 120L * NPTS;  // l0 at channels [120,144) of L1

  // ---- weight prep ----
  const int cCin[3] = {144, 312, 480};
  for (int i = 0; i < 3; ++i)
    k_wprep<<<(48 * cKP[i] + 255) / 256, 256, 0, stream>>>(cw[i], 48, 48,
                                                           cCin[i], cKP[i],
                                                           cWb[i]);
  const int dCin[4][3] = {{48, 72, 96}, {96, 120, 144}, {96, 120, 144},
                          {96, 120, 144}};
  for (int blk = 0; blk < 4; ++blk)
    for (int l = 0; l < 3; ++l)
      k_wprep<<<(32 * dKP[blk][l] + 255) / 256, 256, 0, stream>>>(
          dW[blk][l], 24, 32, dCin[blk][l], dKP[blk][l], dWb[blk][l]);

  dim3 gPts((NB * NPTS) / 256);
  dim3 gT(NPTS / 64, NB);
  dim3 gKnn(NPTS / 16, NB);
  dim3 gDn(NPTS / DB_NT, NB);

  // stage 1
  k_conv_c1<<<gPts, 256, 0, stream>>>(xin, c1w, c1b, l0, s144);
  k_x2bf<<<gT, 256, 0, stream>>>(l0, s144, 24, 32, M0);
  k_sqnorm<<<gPts, 256, 0, stream>>>(l0, s144, 24, SQ);
  k_knn<24, 32><<<gKnn, 512, 0, stream>>>(M0, SQ, IDX);
  k_dense<24, 32><<<gDn, 256, 0, stream>>>(M0, IDX, dWb[0][0], dB[0][0],
                                           dWb[0][1], dB[0][1], dWb[0][2],
                                           dB[0][2], L1, s144);
  // stage 2
  k_x2bf<<<gT, 256, 0, stream>>>(L1, s144, 144, 160, ML1);
  k_conv1x1<160><<<gT, 256, 0, stream>>>(ML1, cWb[0], cb[0], X, s48);
  k_x2bf<<<gT, 256, 0, stream>>>(X, s48, 48, 64, MX);
  k_sqnorm<<<gPts, 256, 0, stream>>>(X, s48, 48, SQ);
  k_knn<48, 64><<<gKnn, 512, 0, stream>>>(MX, SQ, IDX);
  k_copy<<<1024, 256, 0, stream>>>(L1, s144, L2 + 168L * NPTS, s312, s144);
  k_dense<48, 64><<<gDn, 256, 0, stream>>>(MX, IDX, dWb[1][0], dB[1][0],
                                           dWb[1][1], dB[1][1], dWb[1][2],
                                           dB[1][2], L2, s312);
  // stage 3
  k_x2bf<<<gT, 256, 0, stream>>>(L2, s312, 312, 320, ML2);
  k_conv1x1<320><<<gT, 256, 0, stream>>>(ML2, cWb[1], cb[1], X, s48);
  k_x2bf<<<gT, 256, 0, stream>>>(X, s48, 48, 64, MX);
  k_sqnorm<<<gPts, 256, 0, stream>>>(X, s48, 48, SQ);
  k_knn<48, 64><<<gKnn, 512, 0, stream>>>(MX, SQ, IDX);
  k_copy<<<1024, 256, 0, stream>>>(L2, s312, L3 + 168L * NPTS, s480, s312);
  k_dense<48, 64><<<gDn, 256, 0, stream>>>(MX, IDX, dWb[2][0], dB[2][0],
                                           dWb[2][1], dB[2][1], dWb[2][2],
                                           dB[2][2], L3, s480);
  // stage 4
  k_x2bf<<<gT, 256, 0, stream>>>(L3, s480, 480, 480, ML3);
  k_conv1x1<480><<<gT, 256, 0, stream>>>(ML3, cWb[2], cb[2], X, s48);
  k_x2bf<<<gT, 256, 0, stream>>>(X, s48, 48, 64, MX);
  k_sqnorm<<<gPts, 256, 0, stream>>>(X, s48, 48, SQ);
  k_knn<48, 64><<<gKnn, 512, 0, stream>>>(MX, SQ, IDX);
  k_copy<<<1024, 256, 0, stream>>>(L3, s480, out + 168L * NPTS, s648, s480);
  k_dense<48, 64><<<gDn, 256, 0, stream>>>(MX, IDX, dWb[3][0], dB[3][0],
                                           dWb[3][1], dB[3][1], dWb[3][2],
                                           dB[3][2], out, s648);
}
